// CustomAttention_87754771792659
// MI455X (gfx1250) — compile-verified, matrix operand order checked by reading, not run on hardware
//
#include <hip/hip_runtime.h>
#include <hip/hip_bf16.h>

// ---------------------------------------------------------------------------
// Multi-head attention for MI455X (gfx1250, wave32, WMMA).
//   D_MODEL=1024, H=16, d_k=64, S=4096, B derived from input size.
// All matmuls run on v_wmma_f32_16x16x32_f16 (f16 in, f32 accum).
// B-side operands (W rows / K,V tiles) are shared by the 8 waves of a block
// and staged into LDS with double-buffered global_load_async_to_lds_b128
// (ASYNCcnt), so WMMAs consume LDS while the async unit fills the next tile.
// Attention uses the "transposed score" formulation so the softmax
// probabilities land directly in the WMMA B-fragment layout (no LDS
// transpose, no cross-lane permutes except two shfl_xor(16) per step).
// ---------------------------------------------------------------------------

typedef __attribute__((ext_vector_type(16))) _Float16 v16h;
typedef __attribute__((ext_vector_type(8)))  _Float16 v8h;
typedef __attribute__((ext_vector_type(8)))  float    v8f;

#define DMODEL 1024
#define SEQ    4096
#define NHEADS 16
#define DKH    64

#define WAIT_ASYNCCNT(N) asm volatile("s_wait_asynccnt " #N ::: "memory")

// Async DMA of one 16-byte chunk per lane: global -> LDS, tracked by ASYNCcnt.
__device__ __forceinline__ void async_b128(unsigned lds_byte_off,
                                           const void* gaddr) {
  asm volatile("global_load_async_to_lds_b128 %0, %1, off"
               :: "v"(lds_byte_off), "v"(gaddr) : "memory");
}

// Flat shared-aperture pointer: low 32 bits == LDS byte offset (ISA §10.2).
__device__ __forceinline__ unsigned lds_off(const void* p) {
  return (unsigned)(unsigned long long)p;
}

// ---- fragment loader -------------------------------------------------------
// 16-bit A/B fragment (16x32 / 32x16) per the CDNA5 ISA layout:
//   lane 0-15 : row = row0+lane,    k elems = kbase + {0..7, 16..23}
//   lane16-31 : row = row0+lane-16, k elems = kbase + 8 + {0..7, 16..23}
// Two contiguous 16-byte loads per lane (global_load_b128 or ds_load_b128).
__device__ __forceinline__ v16h load_frag16(const _Float16* base,
                                            int row0, int rowstride, int kbase) {
  const int lane = threadIdx.x & 31;
  const int l16  = lane & 15;
  const int hb   = (lane >> 4) & 1;
  const _Float16* p = base + (size_t)(row0 + l16) * rowstride + kbase + hb * 8;
  v8h lo = *(const v8h*)(p);
  v8h hi = *(const v8h*)(p + 16);
  return __builtin_shufflevector(lo, hi, 0,1,2,3,4,5,6,7,8,9,10,11,12,13,14,15);
}

__device__ __forceinline__ v8f wmma16(v16h a, v16h b, v8f c) {
  return __builtin_amdgcn_wmma_f32_16x16x32_f16(false, a, false, b, (short)0, c,
                                                false, false);
}

// ---- fp32 -> f16 conversion ------------------------------------------------
__global__ void mha_cvt_f16_kernel(const float* __restrict__ in,
                                   _Float16* __restrict__ out, int n) {
  int i = blockIdx.x * blockDim.x + threadIdx.x;
  int stride = gridDim.x * blockDim.x;
  for (; i < n; i += stride) out[i] = (_Float16)in[i];
}

// ---- GEMM: C = A(rows x 1024, f16) @ W^T(1024 x 1024, f16) + bias ----------
// Block (8 waves) computes a 128x64 C tile; all waves share the 64-wide W
// tile, staged to LDS via async DMA (double buffered, 4KB per stage: one
// b128 chunk per thread). Wave w handles rows m0 = Mblk*128 + w*16.
//   mode 0: f16 row-major out (Q/K projections)
//   mode 1: f16 per-head transposed out Vt[(b*16+h)*64+d][s] (V projection)
//   mode 2: f32 row-major out (final projection, writes d_out)
__global__ void __launch_bounds__(256)
mha_gemm_kernel(const _Float16* __restrict__ A, const _Float16* __restrict__ W,
                const float* __restrict__ bias, _Float16* __restrict__ outh,
                float* __restrict__ outf, int mrows, int mode) {
  __shared__ _Float16 sW[2][64 * 32];              // 2 x 4KB
  const int nblocks = DMODEL / 64;                 // 16
  const int mblk = blockIdx.x / nblocks;
  const int n0   = (blockIdx.x % nblocks) * 64;
  if (mblk * 128 >= mrows) return;                 // block-uniform guard
  const int tid  = threadIdx.x;
  const int m0   = mblk * 128 + (tid >> 5) * 16;
  const int lane = tid & 31;
  const int l16  = lane & 15;
  const int hb   = (lane >> 4) & 1;

  // staging map: thread t moves W[n0 + t/4][k + (t%4)*8 .. +8)
  const int srow = tid >> 2;
  const int schk = tid & 3;
  const _Float16* wsrc = W + (size_t)(n0 + srow) * DMODEL + schk * 8;
  const unsigned sdst0 = lds_off(&sW[0][srow * 32 + schk * 8]);
  const unsigned sdst1 = lds_off(&sW[1][srow * 32 + schk * 8]);

  async_b128(sdst0, wsrc);                         // prologue: stage k=0

  v8f acc[4] = {v8f{}, v8f{}, v8f{}, v8f{}};
  for (int k = 0; k < DMODEL; k += 32) {
    const int cur = (k >> 5) & 1;
    if (k + 32 < DMODEL) {
      async_b128(cur ? sdst0 : sdst1, wsrc + k + 32);
      WAIT_ASYNCCNT(0x1);                          // current stage landed
    } else {
      WAIT_ASYNCCNT(0x0);
    }
    __syncthreads();
    if (k + 128 < DMODEL)
      __builtin_prefetch(A + (size_t)(m0 + l16) * DMODEL + k + 128, 0, 1);
    v16h a  = load_frag16(A, m0, DMODEL, k);       // per-wave A from global
    const _Float16* sw = sW[cur];
    v16h b0 = load_frag16(sw,  0, 32, 0);          // shared W from LDS
    v16h b1 = load_frag16(sw, 16, 32, 0);
    v16h b2 = load_frag16(sw, 32, 32, 0);
    v16h b3 = load_frag16(sw, 48, 32, 0);
    acc[0] = wmma16(a, b0, acc[0]);
    acc[1] = wmma16(a, b1, acc[1]);
    acc[2] = wmma16(a, b2, acc[2]);
    acc[3] = wmma16(a, b3, acc[3]);
    __syncthreads();                               // reads done before rewrite
  }

  // C/D layout: lane holds n = n0 + nb*16 + l16, component r holds m = m0+r+8*hb
  if (mode == 1) {
    const int gm0   = m0 + 8 * hb;                 // 8 consecutive rows (s dim)
    const int bb    = gm0 >> 12;                   // / SEQ
    const int sbase = gm0 & (SEQ - 1);
    #pragma unroll
    for (int nb = 0; nb < 4; nb++) {
      const int gn = n0 + nb * 16 + l16;
      const int h  = gn >> 6, d = gn & 63;
      const float bv = bias[gn];
      v8h pk;
      #pragma unroll
      for (int r = 0; r < 8; r++) pk[r] = (_Float16)(acc[nb][r] + bv);
      *(v8h*)(outh + (((size_t)(bb * NHEADS + h) * DKH + d) * SEQ + sbase)) = pk;
    }
  } else if (mode == 0) {
    #pragma unroll
    for (int nb = 0; nb < 4; nb++) {
      const int gn = n0 + nb * 16 + l16;
      const float bv = bias[gn];
      #pragma unroll
      for (int r = 0; r < 8; r++) {
        const int gm = m0 + r + 8 * hb;
        outh[(size_t)gm * DMODEL + gn] = (_Float16)(acc[nb][r] + bv);
      }
    }
  } else {
    #pragma unroll
    for (int nb = 0; nb < 4; nb++) {
      const int gn = n0 + nb * 16 + l16;
      const float bv = bias[gn];
      #pragma unroll
      for (int r = 0; r < 8; r++) {
        const int gm = m0 + r + 8 * hb;
        outf[(size_t)gm * DMODEL + gn] = acc[nb][r] + bv;
      }
    }
  }
}

// ---- flash attention (transposed formulation, LDS-shared K/V) --------------
// Block (8 waves) owns (b, h, 128 queries); wave w owns queries q0 = .. + w*16.
// Per 32-key step the block's shared K tile (32x64, 4KB) and V^T tile
// (64x32, 4KB) are async-DMA'd into double-buffered LDS (2 chunks/thread).
//   S^T tiles (2x): A = K rows (16x32 over d), B = Q^T   -> 4 WMMAs over d=64
//   O^T accum    : A = V^T rows (16d x 32k),   B = P^T   -> 4 WMMAs
// The S^T D-tile register layout IS the B-fragment layout of P^T, so the
// exponentiated scores feed the second WMMA with no data movement.
__global__ void __launch_bounds__(256)
mha_attn_kernel(const _Float16* __restrict__ Qm, const _Float16* __restrict__ Km,
                const _Float16* __restrict__ Vt, _Float16* __restrict__ ctx,
                int Bsz) {
  __shared__ _Float16 sK[2][32 * 64];              // 2 x 4KB, row = key
  __shared__ _Float16 sV[2][64 * 32];              // 2 x 4KB, row = d
  const int qblocks = SEQ / 128;                   // 32
  const int qb = blockIdx.x % qblocks;
  const int h  = (blockIdx.x / qblocks) % NHEADS;
  const int b  = blockIdx.x / (qblocks * NHEADS);
  const int tid  = threadIdx.x;
  const int q0   = qb * 128 + (tid >> 5) * 16;
  const int lane = tid & 31;
  const int l16  = lane & 15;
  const int hb   = (lane >> 4) & 1;

  const _Float16* qbase = Qm + (size_t)b * SEQ * DMODEL + h * DKH;
  const _Float16* kbase = Km + (size_t)b * SEQ * DMODEL + h * DKH;
  const _Float16* vbase = Vt + (size_t)(b * NHEADS + h) * DKH * SEQ;

  // staging maps (one b128 chunk per thread per tile)
  const int krow = tid >> 3, kchk = tid & 7;       // K: 32 rows x 128B
  const _Float16* ksrc = kbase + (size_t)krow * DMODEL + kchk * 8;
  const unsigned kdst0 = lds_off(&sK[0][krow * 64 + kchk * 8]);
  const unsigned kdst1 = lds_off(&sK[1][krow * 64 + kchk * 8]);
  const int vrow = tid >> 2, vchk = tid & 3;       // V^T: 64 rows x 64B
  const _Float16* vsrc = vbase + (size_t)vrow * SEQ + vchk * 8;
  const unsigned vdst0 = lds_off(&sV[0][vrow * 32 + vchk * 8]);
  const unsigned vdst1 = lds_off(&sV[1][vrow * 32 + vchk * 8]);

  // loop-invariant Q^T fragments (d = 0..31 and 32..63)
  const v16h qf0 = load_frag16(qbase, q0, DMODEL, 0);
  const v16h qf1 = load_frag16(qbase, q0, DMODEL, 32);

  async_b128(kdst0, ksrc);                         // prologue: stage kk=0
  async_b128(vdst0, vsrc);

  v8f o0 = {}, o1 = {}, o2 = {}, o3 = {};          // O^T: 64d x 16q
  float mrun = -1e30f, lrun = 0.f;
  const float scale = 0.125f;                      // 1/sqrt(64)

  for (int kk = 0; kk < SEQ; kk += 32) {
    const int cur = (kk >> 5) & 1;
    if (kk + 32 < SEQ) {
      async_b128(cur ? kdst0 : kdst1, ksrc + (size_t)(kk + 32) * DMODEL);
      async_b128(cur ? vdst0 : vdst1, vsrc + kk + 32);
      WAIT_ASYNCCNT(0x2);                          // current stage landed
    } else {
      WAIT_ASYNCCNT(0x0);
    }
    __syncthreads();
    const _Float16* skc = sK[cur];
    const _Float16* svc = sV[cur];

    v16h kf00 = load_frag16(skc,  0, 64, 0);
    v16h kf01 = load_frag16(skc,  0, 64, 32);
    v16h kf10 = load_frag16(skc, 16, 64, 0);
    v16h kf11 = load_frag16(skc, 16, 64, 32);
    v8f s0 = {}, s1 = {};                          // S^T: m=key, n=query
    s0 = wmma16(kf00, qf0, s0);
    s0 = wmma16(kf01, qf1, s0);
    s1 = wmma16(kf10, qf0, s1);
    s1 = wmma16(kf11, qf1, s1);

    // online softmax per query (= per lane column, one shfl to merge halves)
    float t0[8], t1[8];
    float mloc = -1e30f;
    #pragma unroll
    for (int r = 0; r < 8; r++) {
      t0[r] = s0[r] * scale;
      t1[r] = s1[r] * scale;
      mloc = fmaxf(mloc, fmaxf(t0[r], t1[r]));
    }
    mloc = fmaxf(mloc, __shfl_xor(mloc, 16, 32));
    const float mnew = fmaxf(mrun, mloc);
    const float corr = __expf(mrun - mnew);
    float psum = 0.f;
    v16h pf;                                       // P^T B-fragment, in place
    #pragma unroll
    for (int r = 0; r < 8; r++) {
      const float p0 = __expf(t0[r] - mnew);
      const float p1 = __expf(t1[r] - mnew);
      psum += p0 + p1;
      pf[r]     = (_Float16)p0;
      pf[r + 8] = (_Float16)p1;
    }
    psum += __shfl_xor(psum, 16, 32);
    lrun = lrun * corr + psum;
    mrun = mnew;
    #pragma unroll
    for (int r = 0; r < 8; r++) {
      o0[r] *= corr; o1[r] *= corr; o2[r] *= corr; o3[r] *= corr;
    }
    v16h vf0 = load_frag16(svc,  0, 32, 0);
    v16h vf1 = load_frag16(svc, 16, 32, 0);
    v16h vf2 = load_frag16(svc, 32, 32, 0);
    v16h vf3 = load_frag16(svc, 48, 32, 0);
    o0 = wmma16(vf0, pf, o0);
    o1 = wmma16(vf1, pf, o1);
    o2 = wmma16(vf2, pf, o2);
    o3 = wmma16(vf3, pf, o3);
    __syncthreads();                               // reads done before rewrite
  }

  // normalize and store: lane's query = q0+l16; component r -> d = db*16+r+8*hb
  const float inv = 1.f / lrun;
  _Float16* crow = ctx + (size_t)(b * SEQ + q0 + l16) * DMODEL + h * DKH + 8 * hb;
  v8f oall[4] = {o0, o1, o2, o3};
  #pragma unroll
  for (int db = 0; db < 4; db++) {
    v8h pk;
    #pragma unroll
    for (int r = 0; r < 8; r++) pk[r] = (_Float16)(oall[db][r] * inv);
    *(v8h*)(crow + db * 16) = pk;
  }
}

// ---------------------------------------------------------------------------
extern "C" void kernel_launch(void* const* d_in, const int* in_sizes, int n_in,
                              void* d_out, int out_size, void* d_ws, size_t ws_size,
                              hipStream_t stream) {
  const float* q  = (const float*)d_in[0];
  const float* k  = (const float*)d_in[1];
  const float* v  = (const float*)d_in[2];
  const float* Wq = (const float*)d_in[3];
  const float* bq = (const float*)d_in[4];
  const float* Wk = (const float*)d_in[5];
  const float* bk = (const float*)d_in[6];
  const float* Wv = (const float*)d_in[7];
  const float* bv = (const float*)d_in[8];
  const float* Wo = (const float*)d_in[9];
  const float* bo = (const float*)d_in[10];

  const int rows = in_sizes[0] / DMODEL;   // B * S
  const int Bsz  = rows / SEQ;

  char* ws = (char*)d_ws;
  const size_t fbytes = (size_t)rows * DMODEL * sizeof(_Float16);   // 16 MB
  const size_t wbytes = (size_t)DMODEL * DMODEL * sizeof(_Float16); //  2 MB
  _Float16* Xq  = (_Float16*)(ws + 0 * fbytes);
  _Float16* Xk  = (_Float16*)(ws + 1 * fbytes);
  _Float16* Xv  = (_Float16*)(ws + 2 * fbytes);
  _Float16* Qf  = (_Float16*)(ws + 3 * fbytes);
  _Float16* Kf  = (_Float16*)(ws + 4 * fbytes);
  _Float16* Vtf = (_Float16*)(ws + 5 * fbytes);
  _Float16* Cf  = (_Float16*)(ws + 6 * fbytes);
  _Float16* wq16 = (_Float16*)(ws + 7 * fbytes + 0 * wbytes);
  _Float16* wk16 = (_Float16*)(ws + 7 * fbytes + 1 * wbytes);
  _Float16* wv16 = (_Float16*)(ws + 7 * fbytes + 2 * wbytes);
  _Float16* wo16 = (_Float16*)(ws + 7 * fbytes + 3 * wbytes);

  const int nX = rows * DMODEL;
  const int nW = DMODEL * DMODEL;
  mha_cvt_f16_kernel<<<2048, 256, 0, stream>>>(q,  Xq,   nX);
  mha_cvt_f16_kernel<<<2048, 256, 0, stream>>>(k,  Xk,   nX);
  mha_cvt_f16_kernel<<<2048, 256, 0, stream>>>(v,  Xv,   nX);
  mha_cvt_f16_kernel<<< 512, 256, 0, stream>>>(Wq, wq16, nW);
  mha_cvt_f16_kernel<<< 512, 256, 0, stream>>>(Wk, wk16, nW);
  mha_cvt_f16_kernel<<< 512, 256, 0, stream>>>(Wv, wv16, nW);
  mha_cvt_f16_kernel<<< 512, 256, 0, stream>>>(Wo, wo16, nW);

  const int gemm_blocks = (rows / 128) * (DMODEL / 64);   // 128x64 tile / block
  mha_gemm_kernel<<<gemm_blocks, 256, 0, stream>>>(Xq, wq16, bq, Qf,  nullptr, rows, 0);
  mha_gemm_kernel<<<gemm_blocks, 256, 0, stream>>>(Xk, wk16, bk, Kf,  nullptr, rows, 0);
  mha_gemm_kernel<<<gemm_blocks, 256, 0, stream>>>(Xv, wv16, bv, Vtf, nullptr, rows, 1);

  const int at_blocks = Bsz * NHEADS * (SEQ / 128);       // 128 queries / block
  mha_attn_kernel<<<at_blocks, 256, 0, stream>>>(Qf, Kf, Vtf, Cf, Bsz);

  mha_gemm_kernel<<<gemm_blocks, 256, 0, stream>>>(Cf, wo16, bo, nullptr,
                                                   (float*)d_out, rows, 2);
}
